// MultiHeadAttention_28956669510170
// MI455X (gfx1250) — compile-verified, matrix operand order checked by reading, not run on hardware
//
#include <hip/hip_runtime.h>
#include <hip/hip_bf16.h>

#define D_MODEL 1024
#define NUM_HEADS 16
#define D_K 64
#define BATCH 4
#define SEQ 2048
#define MROWS (BATCH * SEQ)   // 8192

typedef __attribute__((ext_vector_type(16))) __bf16 v16bf;
typedef __attribute__((ext_vector_type(8)))  __bf16 v8bf;
typedef __attribute__((ext_vector_type(8)))  float  v8f;
typedef __attribute__((ext_vector_type(4)))  float  f4;

__device__ inline __bf16 f2bf(float f) {
    unsigned u = __builtin_bit_cast(unsigned, f);
    u += 0x7FFFu + ((u >> 16) & 1u);            // round-to-nearest-even
    unsigned short h = (unsigned short)(u >> 16);
    return __builtin_bit_cast(__bf16, h);
}

// Fragment loader: A/B 16-bit WMMA layout — lane = (half h, idx lm); elements
// 0..7 cover K = k0 + h*8 + [0..7], elements 8..15 cover K = k0+16+h*8+[0..7].
// Both halves are contiguous 16-byte runs -> two b128 loads.
__device__ inline v16bf load_frag(const __bf16* p) {
    v8bf lo = *(const v8bf*)p;
    v8bf hi = *(const v8bf*)(p + 16);
    v16bf r;
#pragma unroll
    for (int i = 0; i < 8; ++i) { r[i] = lo[i]; r[i + 8] = hi[i]; }
    return r;
}

__device__ inline v8f bfwmma(v16bf a, v16bf b, v8f c) {
    return __builtin_amdgcn_wmma_f32_16x16x32_bf16(false, a, false, b,
                                                   (short)0, c, false, false);
}

// ---------------------------------------------------------------------------
// fp32 -> bf16 elementwise convert (activations), fully vectorized.
// ---------------------------------------------------------------------------
__global__ __launch_bounds__(256) void convert_bf16(const float* __restrict__ X,
                                                    __bf16* __restrict__ Y) {
    size_t idx = ((size_t)blockIdx.x * 256 + threadIdx.x) * 8;
    f4 a = *(const f4*)(X + idx);
    f4 b = *(const f4*)(X + idx + 4);
    v8bf o;
#pragma unroll
    for (int i = 0; i < 4; ++i) { o[i] = f2bf(a[i]); o[i + 4] = f2bf(b[i]); }
    *(v8bf*)(Y + idx) = o;
}

// ---------------------------------------------------------------------------
// Weight transpose + fp32 -> bf16 convert:  Wt[n,k] = bf16(W[k,n])
// ---------------------------------------------------------------------------
__global__ void transpose_convert_w(const float* __restrict__ W,
                                    __bf16* __restrict__ Wt) {
    int idx = blockIdx.x * 256 + threadIdx.x;      // over 1024*1024
    int n = idx & (D_MODEL - 1);
    int k = idx >> 10;
    Wt[(size_t)n * D_MODEL + k] = f2bf(W[idx]);
}

// ---------------------------------------------------------------------------
// Tiled bf16 transpose per batch:  dst[b, c, s] = src[b, s, c]
// 32x32 LDS tiles, coalesced on both sides.
// ---------------------------------------------------------------------------
__global__ __launch_bounds__(256) void transpose_bf16_bsc(
    const __bf16* __restrict__ src, __bf16* __restrict__ dst) {
    __shared__ __bf16 t[32][33];
    const int b = blockIdx.z;
    const int c0 = blockIdx.x * 32;
    const int s0 = blockIdx.y * 32;
    const __bf16* s = src + (size_t)b * SEQ * D_MODEL;
    __bf16* d = dst + (size_t)b * D_MODEL * SEQ;
    const int x = threadIdx.x;                     // 0..31
#pragma unroll
    for (int yy = 0; yy < 32; yy += 8) {
        const int y = threadIdx.y + yy;
        t[y][x] = s[(size_t)(s0 + y) * D_MODEL + c0 + x];
    }
    __syncthreads();
#pragma unroll
    for (int yy = 0; yy < 32; yy += 8) {
        const int y = threadIdx.y + yy;
        d[(size_t)(c0 + y) * SEQ + s0 + x] = t[x][y];
    }
}

// ---------------------------------------------------------------------------
// GEMM:  C[m,n] = sum_k A[m,k] * Bt[n,k] + bias[n]   (A, Bt bf16)
// Block = 128 threads (4 waves), block tile 128x128, wave tile 64x64
// (4x4 WMMA 16x16 tiles), K stepped by 32, double-buffered fragments so the
// WMMA burst on k0 never waits on the k0+32 loads.
// ---------------------------------------------------------------------------
template <bool F32OUT>
__global__ __launch_bounds__(128) void gemm_bias_wmma(
    const __bf16* __restrict__ A, const __bf16* __restrict__ Bt,
    const float* __restrict__ bias, void* __restrict__ Cout,
    int M, int N, int K) {
    const int lane = threadIdx.x & 31;
    const int wave = threadIdx.x >> 5;
    const int half = lane >> 4;
    const int lm   = lane & 15;
    const int m0 = blockIdx.y * 128 + (wave >> 1) * 64;
    const int n0 = blockIdx.x * 128 + (wave & 1) * 64;

    const __bf16* Ab = A + (size_t)(m0 + lm) * K + half * 8;
    const __bf16* Bb = Bt + (size_t)(n0 + lm) * K + half * 8;

    v8f acc[4][4] = {};
    v16bf af[2][4], bfr[2][4];
#pragma unroll
    for (int mi = 0; mi < 4; ++mi) af[0][mi] = load_frag(Ab + (size_t)mi * 16 * K);
#pragma unroll
    for (int ni = 0; ni < 4; ++ni) bfr[0][ni] = load_frag(Bb + (size_t)ni * 16 * K);

    for (int k0 = 0; k0 < K; k0 += 32) {
        const int cur = (k0 >> 5) & 1;
        const int nxt = cur ^ 1;
        if (k0 + 32 < K) {
#pragma unroll
            for (int mi = 0; mi < 4; ++mi)
                af[nxt][mi] = load_frag(Ab + (size_t)mi * 16 * K + k0 + 32);
#pragma unroll
            for (int ni = 0; ni < 4; ++ni)
                bfr[nxt][ni] = load_frag(Bb + (size_t)ni * 16 * K + k0 + 32);
        }
#pragma unroll
        for (int mi = 0; mi < 4; ++mi)
#pragma unroll
            for (int ni = 0; ni < 4; ++ni)
                acc[mi][ni] = bfwmma(af[cur][mi], bfr[cur][ni], acc[mi][ni]);
    }
    // C/D layout: vgpr r, lanes0-15 -> row r, lanes16-31 -> row r+8; col = lm.
#pragma unroll
    for (int ni = 0; ni < 4; ++ni) {
        const int col = n0 + ni * 16 + lm;
        const float bv = bias[col];
#pragma unroll
        for (int mi = 0; mi < 4; ++mi) {
#pragma unroll
            for (int r = 0; r < 8; ++r) {
                const int row = m0 + mi * 16 + r + half * 8;
                const float v = acc[mi][ni][r] + bv;
                if (F32OUT)
                    ((float*)Cout)[(size_t)row * N + col] = v;
                else
                    ((__bf16*)Cout)[(size_t)row * N + col] = f2bf(v);
            }
        }
    }
}

// ---------------------------------------------------------------------------
// Flash attention: block = 4 waves, 128 query rows per block, one (b,h).
// Wave w owns query rows [w*32, w*32+32) (2 M-tiles) — every K/V fragment is
// reused by both M-tiles (32 WMMAs per 16 fragment loads). Waves are fully
// independent (LDS P region is wave-private; LDS is in-order within a wave),
// so no barriers. K and pre-transposed V fragments stream from global/L2.
// ---------------------------------------------------------------------------
__global__ __launch_bounds__(128) void attention_wmma(
    const __bf16* __restrict__ Qb, const __bf16* __restrict__ Kb,
    const __bf16* __restrict__ Vt, __bf16* __restrict__ AO) {
    __shared__ __align__(16) __bf16 ldsP[128][72];   // P: [qrow][key], wave-private rows

    const int lane = threadIdx.x & 31;
    const int wave = threadIdx.x >> 5;
    const int half = lane >> 4;
    const int lm   = lane & 15;
    const int b = blockIdx.y >> 4;
    const int h = blockIdx.y & 15;
    const int q0 = blockIdx.x * 128;

    const __bf16* Qbase = Qb + ((size_t)(b * SEQ + q0)) * D_MODEL + h * D_K;
    const __bf16* Kbase = Kb + (size_t)b * SEQ * D_MODEL + h * D_K;
    // Vt layout: [b, c=(h*64+d), s]
    const __bf16* Vtbase = Vt + ((size_t)b * D_MODEL + h * D_K) * SEQ;

    // Q fragments: 2 M-tiles x 2 K-steps, held in registers the whole kernel.
    v16bf qa[2][2];
#pragma unroll
    for (int mi = 0; mi < 2; ++mi)
#pragma unroll
        for (int kk = 0; kk < 2; ++kk)
            qa[mi][kk] = load_frag(Qbase + (size_t)(wave * 32 + mi * 16 + lm) * D_MODEL +
                                   kk * 32 + half * 8);

    float mrun[2][8], lrun[2][8];
#pragma unroll
    for (int mi = 0; mi < 2; ++mi)
#pragma unroll
        for (int r = 0; r < 8; ++r) { mrun[mi][r] = -3.0e38f; lrun[mi][r] = 0.0f; }
    v8f oacc[2][4] = {};

    for (int j = 0; j < SEQ / 64; ++j) {
        // Prefetch next key tile rows into the cache hierarchy.
        if (j + 1 < SEQ / 64) {
            __builtin_prefetch(Kbase + (size_t)((j + 1) * 64 + (lane << 1)) * D_MODEL, 0, 1);
            __builtin_prefetch(Vtbase + (size_t)(lane << 1) * SEQ + (j + 1) * 64, 0, 1);
        }

        // S = Q * K^T : load the 8 K fragments once, reuse for both M-tiles.
        v16bf kf[2][4];
#pragma unroll
        for (int kk = 0; kk < 2; ++kk)
#pragma unroll
            for (int ni = 0; ni < 4; ++ni)
                kf[kk][ni] = load_frag(Kbase + (size_t)(j * 64 + ni * 16 + lm) * D_MODEL +
                                       kk * 32 + half * 8);
        v8f sacc[2][4] = {};
#pragma unroll
        for (int kk = 0; kk < 2; ++kk)
#pragma unroll
            for (int mi = 0; mi < 2; ++mi)
#pragma unroll
                for (int ni = 0; ni < 4; ++ni)
                    sacc[mi][ni] = bfwmma(qa[mi][kk], kf[kk][ni], sacc[mi][ni]);

        // Online softmax; row = (r + 8*half) lives across the 16 lanes of a half.
#pragma unroll
        for (int mi = 0; mi < 2; ++mi) {
#pragma unroll
            for (int r = 0; r < 8; ++r) {
                float s0 = sacc[mi][0][r] * 0.125f;
                float s1 = sacc[mi][1][r] * 0.125f;
                float s2 = sacc[mi][2][r] * 0.125f;
                float s3 = sacc[mi][3][r] * 0.125f;
                float rm = fmaxf(fmaxf(s0, s1), fmaxf(s2, s3));
                rm = fmaxf(rm, __shfl_xor(rm, 1, 32));
                rm = fmaxf(rm, __shfl_xor(rm, 2, 32));
                rm = fmaxf(rm, __shfl_xor(rm, 4, 32));
                rm = fmaxf(rm, __shfl_xor(rm, 8, 32));
                float mnew = fmaxf(mrun[mi][r], rm);
                float corr = __expf(mrun[mi][r] - mnew);
                float p0 = __expf(s0 - mnew);
                float p1 = __expf(s1 - mnew);
                float p2 = __expf(s2 - mnew);
                float p3 = __expf(s3 - mnew);
                float rs = p0 + p1 + p2 + p3;
                rs += __shfl_xor(rs, 1, 32);
                rs += __shfl_xor(rs, 2, 32);
                rs += __shfl_xor(rs, 4, 32);
                rs += __shfl_xor(rs, 8, 32);
                lrun[mi][r] = lrun[mi][r] * corr + rs;
                mrun[mi][r] = mnew;
#pragma unroll
                for (int ni = 0; ni < 4; ++ni) oacc[mi][ni][r] *= corr;
                const int qrow = wave * 32 + mi * 16 + r + half * 8;  // wave-private rows
                ldsP[qrow][0 * 16 + lm] = f2bf(p0);
                ldsP[qrow][1 * 16 + lm] = f2bf(p1);
                ldsP[qrow][2 * 16 + lm] = f2bf(p2);
                ldsP[qrow][3 * 16 + lm] = f2bf(p3);
            }
        }

        // O += P * V : load the 4 V fragments per K-step once, reuse both M-tiles.
#pragma unroll
        for (int kk = 0; kk < 2; ++kk) {
            v16bf vf[4];
#pragma unroll
            for (int ni = 0; ni < 4; ++ni)
                vf[ni] = load_frag(Vtbase + (size_t)(ni * 16 + lm) * SEQ +
                                   j * 64 + kk * 32 + half * 8);
#pragma unroll
            for (int mi = 0; mi < 2; ++mi) {
                v16bf pf = load_frag(
                    (const __bf16*)&ldsP[wave * 32 + mi * 16 + lm][kk * 32 + half * 8]);
#pragma unroll
                for (int ni = 0; ni < 4; ++ni)
                    oacc[mi][ni] = bfwmma(pf, vf[ni], oacc[mi][ni]);
            }
        }
    }

    // Normalize and write [b, q, h*64+d] so the Wo GEMM reads it row-major.
#pragma unroll
    for (int mi = 0; mi < 2; ++mi) {
#pragma unroll
        for (int ni = 0; ni < 4; ++ni) {
#pragma unroll
            for (int r = 0; r < 8; ++r) {
                const int qrow = q0 + wave * 32 + mi * 16 + r + half * 8;
                const int col = h * D_K + ni * 16 + lm;
                AO[((size_t)b * SEQ + qrow) * D_MODEL + col] =
                    f2bf(oacc[mi][ni][r] / lrun[mi][r]);
            }
        }
    }
}

// ---------------------------------------------------------------------------
extern "C" void kernel_launch(void* const* d_in, const int* in_sizes, int n_in,
                              void* d_out, int out_size, void* d_ws, size_t ws_size,
                              hipStream_t stream) {
    const float* query = (const float*)d_in[0];
    const float* key_  = (const float*)d_in[1];
    const float* value = (const float*)d_in[2];
    const float* Wq = (const float*)d_in[3];
    const float* bq = (const float*)d_in[4];
    const float* Wk = (const float*)d_in[5];
    const float* bk = (const float*)d_in[6];
    const float* Wv = (const float*)d_in[7];
    const float* bv = (const float*)d_in[8];
    const float* Wo = (const float*)d_in[9];
    const float* bo = (const float*)d_in[10];
    float* out = (float*)d_out;

    char* ws = (char*)d_ws;
    const size_t MB = 1024ull * 1024ull;
    __bf16* WqT = (__bf16*)(ws + 0 * MB);
    __bf16* WkT = (__bf16*)(ws + 2 * MB);
    __bf16* WvT = (__bf16*)(ws + 4 * MB);
    __bf16* WoT = (__bf16*)(ws + 6 * MB);
    __bf16* Xq  = (__bf16*)(ws + 8 * MB);    // 16 MB; later reused as AO
    __bf16* Xk  = (__bf16*)(ws + 24 * MB);   // 16 MB; later reused as Vt
    __bf16* Xv  = (__bf16*)(ws + 40 * MB);   // 16 MB
    __bf16* Qb  = (__bf16*)(ws + 56 * MB);
    __bf16* Kb  = (__bf16*)(ws + 72 * MB);
    __bf16* Vb  = (__bf16*)(ws + 88 * MB);   // total 104 MB
    __bf16* Vtr = Xk;                        // alias: Xk consumed before Vt is written
    __bf16* AO  = Xq;                        // alias: Xq consumed before AO is written

    const int wgrid = (D_MODEL * D_MODEL) / 256;
    transpose_convert_w<<<wgrid, 256, 0, stream>>>(Wq, WqT);
    transpose_convert_w<<<wgrid, 256, 0, stream>>>(Wk, WkT);
    transpose_convert_w<<<wgrid, 256, 0, stream>>>(Wv, WvT);
    transpose_convert_w<<<wgrid, 256, 0, stream>>>(Wo, WoT);

    const int cgrid = (MROWS * D_MODEL) / (256 * 8);
    convert_bf16<<<cgrid, 256, 0, stream>>>(query, Xq);
    convert_bf16<<<cgrid, 256, 0, stream>>>(key_, Xk);
    convert_bf16<<<cgrid, 256, 0, stream>>>(value, Xv);

    dim3 ggrid(D_MODEL / 128, MROWS / 128);   // (8, 64)
    gemm_bias_wmma<false><<<ggrid, 128, 0, stream>>>(Xq, WqT, bq, Qb,
                                                     MROWS, D_MODEL, D_MODEL);
    gemm_bias_wmma<false><<<ggrid, 128, 0, stream>>>(Xk, WkT, bk, Kb,
                                                     MROWS, D_MODEL, D_MODEL);
    gemm_bias_wmma<false><<<ggrid, 128, 0, stream>>>(Xv, WvT, bv, Vb,
                                                     MROWS, D_MODEL, D_MODEL);

    dim3 tgrid(D_MODEL / 32, SEQ / 32, BATCH);
    transpose_bf16_bsc<<<tgrid, dim3(32, 8), 0, stream>>>(Vb, Vtr);

    dim3 agrid(SEQ / 128, BATCH * NUM_HEADS); // (16, 64)
    attention_wmma<<<agrid, 128, 0, stream>>>(Qb, Kb, Vtr, AO);

    gemm_bias_wmma<true><<<ggrid, 128, 0, stream>>>(AO, WoT, bo, out,
                                                    MROWS, D_MODEL, D_MODEL);
}
